// Grapher_33947421508470
// MI455X (gfx1250) — matrix operand order repaired and checked by reading, not compiled, not run on hardware
//
#include <hip/hip_runtime.h>

// ---------------------------------------------------------------------------
// Grapher block (Vision-GNN MRConv2d grapher) for MI455X / gfx1250.
// B=8, C=96, H=W=56, N=3136, K=9 neighbors, G=4 groups.
// Dominant cost: 8 batched Gram matrices 3136x96x3136 (15.1 GFLOP) + top-9.
// Strategy: bf16 WMMA (v_wmma_f32_16x16x32_bf16) for the Gram matrix with
// operands pre-packed into the ISA fragment layouts; VALU for the small 1x1
// convs; everything stays L2-resident (~20 MB workspace << 192 MB L2).
// ---------------------------------------------------------------------------

typedef __attribute__((ext_vector_type(16))) __bf16 v16bf;
typedef __attribute__((ext_vector_type(8)))  float  v8f;

#define BATCH 8
#define CCH   96
#define NPIX  3136          // 56*56
#define NT    196           // NPIX / 16 column/row tiles
#define KNN   9
#define EPSBN 1e-5f

// workspace layout (bytes, all naturally 256-aligned)
#define OFF_FEAT   ((size_t)0)                       // 8*96*3136 f32   = 9,633,792 B
#define OFF_SQ     ((size_t)9633792)                 // 8*3136 f32      =   100,352 B
#define OFF_IDX    ((size_t)9734144)                 // 8*3136*9 i32    =   903,168 B
#define OFF_APACK  ((size_t)10637312)                // 8*196*3*512 bf16= 4,816,896 B
#define OFF_BPACK  ((size_t)15454208)                // same            = 4,816,896 B

__device__ __forceinline__ unsigned short f2bf(float f) {
    unsigned int u = __float_as_uint(f);
    unsigned int r = u + 0x7FFFu + ((u >> 16) & 1u);   // round-to-nearest-even
    return (unsigned short)(r >> 16);
}

// ---------------------------------------------------------------------------
// Kernel 1: fc1 (1x1 conv) + BN1, L2-normalize over channels, emit:
//   feat[b][c][n] (f32), sq[b][n] (f32, sum of squares of normalized vec),
//   bf16 normalized features pre-swizzled into WMMA A- and B-fragment layouts.
// Block = 96 threads (one per channel), 8 pixels per block.
// ---------------------------------------------------------------------------
__global__ __launch_bounds__(96) void fc1_bn_pack_kernel(
    const float* __restrict__ x,  const float* __restrict__ w,
    const float* __restrict__ bias,
    const float* __restrict__ g1, const float* __restrict__ b1,
    const float* __restrict__ m1, const float* __restrict__ v1,
    float* __restrict__ feat, float* __restrict__ sq,
    unsigned short* __restrict__ apack, unsigned short* __restrict__ bpack)
{
    const int b  = blockIdx.x / (NPIX / 8);
    const int n0 = (blockIdx.x % (NPIX / 8)) * 8;
    const int c  = threadIdx.x;

    __shared__ float xs[CCH][8];
    __shared__ float wred[3];

    const float* xrow = x + ((size_t)b * CCH + c) * NPIX + n0;
#pragma unroll
    for (int p = 0; p < 8; ++p) xs[c][p] = xrow[p];
    __syncthreads();

    float acc[8];
#pragma unroll
    for (int p = 0; p < 8; ++p) acc[p] = 0.f;

    const float* wr = w + (size_t)c * CCH;
    for (int k = 0; k < CCH; ++k) {
        float wv = wr[k];
        float4 xa = *(const float4*)&xs[k][0];
        float4 xb = *(const float4*)&xs[k][4];
        acc[0] += wv * xa.x; acc[1] += wv * xa.y;
        acc[2] += wv * xa.z; acc[3] += wv * xa.w;
        acc[4] += wv * xb.x; acc[5] += wv * xb.y;
        acc[6] += wv * xb.z; acc[7] += wv * xb.w;
    }

    const float inv  = g1[c] * rsqrtf(v1[c] + EPSBN);
    const float beta = b1[c] - m1[c] * inv;
    const float bia  = bias[c];

    const int kb = c >> 5, K2 = c & 31;
    for (int p = 0; p < 8; ++p) {
        float h = (acc[p] + bia) * inv + beta;

        // block-wide sum of h^2 (3 waves of 32)
        float ss = h * h;
#pragma unroll
        for (int off = 16; off > 0; off >>= 1) ss += __shfl_xor(ss, off, 32);
        if ((c & 31) == 0) wred[c >> 5] = ss;
        __syncthreads();
        float tot = wred[0] + wred[1] + wred[2];
        __syncthreads();

        float rinv = 1.0f / fmaxf(sqrtf(tot), 1e-12f);
        const int n = n0 + p;

        feat[((size_t)b * CCH + c) * NPIX + n] = h;
        if (c == 0) sq[(size_t)b * NPIX + n] = tot * rinv * rinv;

        unsigned short bf = f2bf(h * rinv);
        const int t = n >> 4, rloc = n & 15;

        // B fragment (32x16, K x col): lane = ((K2>>3)&1)*16 + col, slot = (K2&7)|((K2>>4)<<3)
        {
            int laneB = (((K2 >> 3) & 1) << 4) | rloc;
            int slotB = (K2 & 7) | ((K2 >> 4) << 3);
            size_t ib = ((((size_t)b * NT + t) * 3 + kb) * 32 + laneB) * 16 + slotB;
            bpack[ib] = bf;
        }
        // A fragment (16x32, M x K): lane-half h, slot s per ISA table
        {
            int hA = (K2 >> 3) & 1;
            int sA = (K2 & 7) | ((K2 >> 4) << 3);
            int laneA = (hA << 4) | rloc;
            size_t ia = ((((size_t)b * NT + t) * 3 + kb) * 32 + laneA) * 16 + sA;
            apack[ia] = bf;
        }
    }
}

// ---------------------------------------------------------------------------
// Kernel 2: dynamic-KNN via WMMA Gram matrix + per-lane top-9.
// One wave (32 threads) per 32 query rows: 2 A-tiles held in registers,
// sq row preloaded to LDS once, loop over 196 column tiles:
// 3 B loads + 6 v_wmma_f32_16x16x32_bf16, LDS redistribute so lane L owns
// query row (g*32+L), branchless sorted top-9 insert (cndmask chain).
// ---------------------------------------------------------------------------
__global__ __launch_bounds__(32) void knn_kernel(
    const unsigned short* __restrict__ apacku,
    const unsigned short* __restrict__ bpacku,
    const float* __restrict__ sq, int* __restrict__ idx)
{
    const int b    = blockIdx.x / (NPIX / 32);
    const int g    = blockIdx.x % (NPIX / 32);
    const int lane = threadIdx.x;

    const v16bf* Ap = (const v16bf*)apacku;
    const v16bf* Bp = (const v16bf*)bpacku;

    __shared__ float tile[32][17];   // +1 pad: conflict-free column reads
    __shared__ float sqrow[NPIX];    // whole sq row of this batch (12.5 KB)

    // preload sq row once (coalesced), removes global loads from hot loop
    {
        const float* sqb = sq + (size_t)b * NPIX;
        for (int i = lane; i < NPIX; i += 32) sqrow[i] = sqb[i];
    }

    const size_t abase = ((size_t)b * NT + 2 * g) * 3;
    v16bf a00 = Ap[(abase + 0) * 32 + lane];
    v16bf a01 = Ap[(abase + 1) * 32 + lane];
    v16bf a02 = Ap[(abase + 2) * 32 + lane];
    v16bf a10 = Ap[(abase + 3) * 32 + lane];
    v16bf a11 = Ap[(abase + 4) * 32 + lane];
    v16bf a12 = Ap[(abase + 5) * 32 + lane];

    float vals[9]; int inds[9];
#pragma unroll
    for (int q = 0; q < 9; ++q) { vals[q] = -3.0e38f; inds[q] = 0; }

    const int hrow = (lane >> 4) << 3;   // 0 or 8
    const int colL = lane & 15;

    __syncthreads();

    const size_t bb0 = (size_t)b * NT * 3;
    for (int ct = 0; ct < NT; ++ct) {
        const size_t bbase = bb0 + (size_t)ct * 3;
        v16bf b0 = Bp[(bbase + 0) * 32 + lane];
        v16bf b1 = Bp[(bbase + 1) * 32 + lane];
        v16bf b2 = Bp[(bbase + 2) * 32 + lane];

        // prefetch B fragments two tiles ahead (global_prefetch_b8),
        // one cacheline per lane covers the 3 KB tile
        if (ct + 2 < NT) {
            const char* pf = (const char*)(Bp + (bbase + 6) * 32);
            __builtin_prefetch(pf + lane * 96, 0, 1);
        }

        v8f c0 = {0.f,0.f,0.f,0.f,0.f,0.f,0.f,0.f};
        v8f c1 = {0.f,0.f,0.f,0.f,0.f,0.f,0.f,0.f};
        c0 = __builtin_amdgcn_wmma_f32_16x16x32_bf16(false, a00, false, b0, (short)0, c0, false, false);
        c0 = __builtin_amdgcn_wmma_f32_16x16x32_bf16(false, a01, false, b1, (short)0, c0, false, false);
        c0 = __builtin_amdgcn_wmma_f32_16x16x32_bf16(false, a02, false, b2, (short)0, c0, false, false);
        c1 = __builtin_amdgcn_wmma_f32_16x16x32_bf16(false, a10, false, b0, (short)0, c1, false, false);
        c1 = __builtin_amdgcn_wmma_f32_16x16x32_bf16(false, a11, false, b1, (short)0, c1, false, false);
        c1 = __builtin_amdgcn_wmma_f32_16x16x32_bf16(false, a12, false, b2, (short)0, c1, false, false);

#pragma unroll
        for (int v = 0; v < 8; ++v) {
            tile[hrow + v][colL]      = c0[v];
            tile[16 + hrow + v][colL] = c1[v];
        }
        __syncthreads();

        // lane L owns query row g*32+L: scan 16 candidates of this tile
        for (int nn = 0; nn < 16; ++nn) {
            float s = 2.0f * tile[lane][nn] - sqrow[ct * 16 + nn];
            if (s > vals[8]) {                      // rare at steady state
                float cv = s; int ci = ct * 16 + nn;
#pragma unroll
                for (int q = 0; q < 9; ++q) {       // branchless swap chain
                    bool gt  = cv > vals[q];        // strict >: stable ties
                    float tv = vals[q]; int ti = inds[q];
                    vals[q] = gt ? cv : tv;
                    inds[q] = gt ? ci : ti;
                    cv      = gt ? tv : cv;
                    ci      = gt ? ti : ci;
                }
            }
        }
        __syncthreads();
    }

    const int row = g * 32 + lane;
    int* op = idx + ((size_t)b * NPIX + row) * KNN;
#pragma unroll
    for (int q = 0; q < 9; ++q) op[q] = inds[q];
}

// ---------------------------------------------------------------------------
// Kernel 3: gather + max-relative -> interleaved hcat (LDS) -> grouped 1x1
// conv (G=4) + ReLU -> fc2 -> BN2 -> + shortcut. 192 threads, 4 pixels/block.
// ---------------------------------------------------------------------------
__global__ __launch_bounds__(192) void tail_kernel(
    const float* __restrict__ feat, const int* __restrict__ idx,
    const float* __restrict__ gw,   const float* __restrict__ gb,
    const float* __restrict__ w2,   const float* __restrict__ b2v,
    const float* __restrict__ bg,   const float* __restrict__ bb,
    const float* __restrict__ bm,   const float* __restrict__ bv,
    const float* __restrict__ x,    float* __restrict__ out)
{
    const int b  = blockIdx.x / (NPIX / 4);
    const int n0 = (blockIdx.x % (NPIX / 4)) * 4;
    const int t  = threadIdx.x;

    __shared__ float hin[4][2 * CCH];
    __shared__ float ybuf[4][2 * CCH];

    if (t < CCH) {
        const float* fb = feat + ((size_t)b * CCH + t) * NPIX;
#pragma unroll
        for (int p = 0; p < 4; ++p) {
            const int n = n0 + p;
            float fv = fb[n];
            const int* ip = idx + ((size_t)b * NPIX + n) * KNN;
            float m = -3.0e38f;
#pragma unroll
            for (int k = 0; k < KNN; ++k) m = fmaxf(m, fb[ip[k]]);
            hin[p][2 * t]     = fv;        // interleaved [feat, max_rel]
            hin[p][2 * t + 1] = m - fv;
        }
    }
    __syncthreads();

    // grouped 1x1 conv (4 groups of 48) + ReLU
    {
        const int base = (t / 48) * 48;
        const float* wr = gw + (size_t)t * 48;
        float a0 = gb[t], a1 = a0, a2 = a0, a3 = a0;
        for (int i = 0; i < 48; ++i) {
            float wv = wr[i];
            a0 += wv * hin[0][base + i];
            a1 += wv * hin[1][base + i];
            a2 += wv * hin[2][base + i];
            a3 += wv * hin[3][base + i];
        }
        ybuf[0][t] = fmaxf(a0, 0.f);
        ybuf[1][t] = fmaxf(a1, 0.f);
        ybuf[2][t] = fmaxf(a2, 0.f);
        ybuf[3][t] = fmaxf(a3, 0.f);
    }
    __syncthreads();

    // fc2 (192 -> 96) + BN2 + residual
    if (t < CCH) {
        const float* wrow = w2 + (size_t)t * (2 * CCH);
        float z0 = b2v[t], z1 = z0, z2 = z0, z3 = z0;
        for (int o = 0; o < 2 * CCH; ++o) {
            float wv = wrow[o];
            z0 += wv * ybuf[0][o];
            z1 += wv * ybuf[1][o];
            z2 += wv * ybuf[2][o];
            z3 += wv * ybuf[3][o];
        }
        float inv  = bg[t] * rsqrtf(bv[t] + EPSBN);
        float beta = bb[t] - bm[t] * inv;
        const float* xr  = x   + ((size_t)b * CCH + t) * NPIX;
        float*       orw = out + ((size_t)b * CCH + t) * NPIX;
        orw[n0 + 0] = z0 * inv + beta + xr[n0 + 0];
        orw[n0 + 1] = z1 * inv + beta + xr[n0 + 1];
        orw[n0 + 2] = z2 * inv + beta + xr[n0 + 2];
        orw[n0 + 3] = z3 * inv + beta + xr[n0 + 3];
    }
}

// ---------------------------------------------------------------------------
extern "C" void kernel_launch(void* const* d_in, const int* in_sizes, int n_in,
                              void* d_out, int out_size, void* d_ws, size_t ws_size,
                              hipStream_t stream)
{
    const float* x     = (const float*)d_in[0];
    const float* fc1_w = (const float*)d_in[1];
    const float* fc1_b = (const float*)d_in[2];
    const float* bn1_g = (const float*)d_in[3];
    const float* bn1_b = (const float*)d_in[4];
    const float* bn1_m = (const float*)d_in[5];
    const float* bn1_v = (const float*)d_in[6];
    const float* g_w   = (const float*)d_in[7];
    const float* g_b   = (const float*)d_in[8];
    const float* fc2_w = (const float*)d_in[9];
    const float* fc2_b = (const float*)d_in[10];
    const float* bn2_g = (const float*)d_in[11];
    const float* bn2_b = (const float*)d_in[12];
    const float* bn2_m = (const float*)d_in[13];
    const float* bn2_v = (const float*)d_in[14];

    char* ws = (char*)d_ws;
    float*          feat  = (float*)(ws + OFF_FEAT);
    float*          sq    = (float*)(ws + OFF_SQ);
    int*            idx   = (int*)(ws + OFF_IDX);
    unsigned short* apack = (unsigned short*)(ws + OFF_APACK);
    unsigned short* bpack = (unsigned short*)(ws + OFF_BPACK);
    float*          out   = (float*)d_out;

    // 1) fc1 + BN1 + normalize + WMMA-fragment packing
    fc1_bn_pack_kernel<<<dim3(BATCH * (NPIX / 8)), dim3(96), 0, stream>>>(
        x, fc1_w, fc1_b, bn1_g, bn1_b, bn1_m, bn1_v, feat, sq, apack, bpack);

    // 2) WMMA Gram matrix + top-9 KNN
    knn_kernel<<<dim3(BATCH * (NPIX / 32)), dim3(32), 0, stream>>>(
        apack, bpack, sq, idx);

    // 3) max-relative gather + grouped conv + fc2 + BN2 + residual
    tail_kernel<<<dim3(BATCH * (NPIX / 4)), dim3(192), 0, stream>>>(
        feat, idx, g_w, g_b, fc2_w, fc2_b, bn2_g, bn2_b, bn2_m, bn2_v, x, out);
}
